// MultiHeadSelection_79422535238044
// MI455X (gfx1250) — matrix operand order repaired and checked by reading, not compiled, not run on hardware
//
#include <hip/hip_runtime.h>
#include <math.h>

// ---------------- model dims ----------------
#define Bn   16
#define Ln   128
#define En   300
#define EPn  320          // E padded to multiple of 32
#define Hn   512
#define FH   2048         // 4*H
#define Tn   9
#define BIOE 64
#define RELE 128
#define Rn   50
#define RPn  64           // R padded to multiple of 16
#define OCW  576          // H + BIO_E = 18*32
#define BL   (Bn*Ln)      // 2048

typedef __attribute__((ext_vector_type(16))) __bf16 v16bf;
typedef __attribute__((ext_vector_type(8)))  float  v8f;

#define WMMA_BF16(C, A, Bm) \
  (C) = __builtin_amdgcn_wmma_f32_16x16x32_bf16(false, (A), false, (Bm), (short)0, (C), false, false)

// A-fragment (16x32 bf16) from a row-major bf16 row pointer.
// lane gives M = lane&15 (caller bakes it into `row`); hi = lane>>4.
// half index i -> K = k0 + 16*(i/8) + 8*hi + (i%8)
static __device__ __forceinline__ v16bf afrag_bf16(const __bf16* row, int k0, int hi) {
  v16bf a;
  const __bf16* p = row + k0 + hi * 8;
#pragma unroll
  for (int i = 0; i < 8; ++i) a[i] = p[i];
#pragma unroll
  for (int i = 0; i < 8; ++i) a[8 + i] = p[16 + i];
  return a;
}

// B-fragment (32x16 bf16) where Bmat[k][n] = Wb[n][k], Wb pre-packed bf16.
// caller passes row = Wb + n*ld. half index i -> K = k0 + 16*(i/8) + 8*hi + (i%8).
static __device__ __forceinline__ v16bf bfrag_bf16(const __bf16* row, int k0, int hi) {
  v16bf b;
  const __bf16* p = row + k0 + hi * 8;
#pragma unroll
  for (int i = 0; i < 8; ++i) b[i] = p[i];
#pragma unroll
  for (int i = 0; i < 8; ++i) b[8 + i] = p[16 + i];
  return b;
}

// ---------------- kernel 0: generic f32 -> bf16 weight packer --------------
// dst[r*kpad + k] = (r < rin && k < kin) ? src[r*ld + off + k] : 0
__global__ void pack_bf16_kernel(const float* __restrict__ src, int ld, int off,
                                 int rin, int kin, int kpad,
                                 __bf16* __restrict__ dst) {
  int r = blockIdx.x;
  for (int k = threadIdx.x; k < kpad; k += blockDim.x) {
    float v = (r < rin && k < kin) ? src[r * ld + off + k] : 0.0f;
    dst[r * kpad + k] = (__bf16)v;
  }
}

// ---------------- kernel 1: embedding gather + pad to bf16 ----------------
__global__ void embed_kernel(const int* __restrict__ tokens,
                             const float* __restrict__ word_emb,
                             __bf16* __restrict__ emb) {
  int bl  = blockIdx.x;                 // 0..2047
  int tok = tokens[bl];
  for (int e = threadIdx.x; e < EPn; e += blockDim.x) {
    float v = (e < En) ? word_emb[tok * En + e] : 0.0f;
    emb[bl * EPn + e] = (__bf16)v;
  }
}

// ---------------- kernel 2: input projection GEMM  (preact[t][b][j]) -------
// X (2048 x 320 bf16, rows ordered (t,b) with optional time reversal)
// @ w_ih^T (320 x 2048, pre-packed bf16) + (b_ih + b_hh)
__global__ void input_proj_kernel(const __bf16* __restrict__ emb,
                                  const __bf16* __restrict__ w_ih_bf,
                                  const float* __restrict__ b_ih,
                                  const float* __restrict__ b_hh,
                                  float* __restrict__ preact, int rev) {
  int t    = blockIdx.x;               // 0..127 : one 16-row tile = one timestep
  int ng   = blockIdx.y;               // 0..31  : group of 4 n-tiles
  int w    = threadIdx.x >> 5;         // 0..3
  int lane = threadIdx.x & 31;
  int hi   = lane >> 4;

  int src_t = rev ? (Ln - 1 - t) : t;
  const __bf16* arow = emb + ((lane & 15) * Ln + src_t) * EPn;

  int nt = ng * 4 + w;
  int n  = nt * 16 + (lane & 15);
  const __bf16* wrow = w_ih_bf + n * EPn;

  v8f acc = {0.f,0.f,0.f,0.f,0.f,0.f,0.f,0.f};
#pragma unroll
  for (int kk = 0; kk < EPn / 32; ++kk) {
    int k0 = kk * 32;
    v16bf a = afrag_bf16(arow, k0, hi);
    v16bf b = bfrag_bf16(wrow, k0, hi);
    WMMA_BF16(acc, a, b);
  }
  float bias = b_ih[n] + b_hh[n];
#pragma unroll
  for (int g = 0; g < 8; ++g) {
    int brow = g + 8 * hi;             // batch index within the tile
    preact[(t * Bn + brow) * FH + n] = acc[g] + bias;
  }
}

// ---------------- kernel 3: persistent recurrent LSTM (2 blocks) ----------
// block 0 = forward, block 1 = backward; h(bf16)/c(f32) resident in LDS.
// 32 waves; wave w owns 'within' slice [w*16, w*16+16) across all 4 gates.
// w_hh pre-packed bf16: B-frags are pure 2x b128 loads, no cvt on the
// 128-step dependent chain. Next step's preact prefetched over the GEMM.
__global__ void __launch_bounds__(1024, 1)
lstm_rec_kernel(const float* __restrict__ preact_f, const float* __restrict__ preact_b,
                const __bf16* __restrict__ whhb_f,  const __bf16* __restrict__ whhb_b,
                float* __restrict__ hf, float* __restrict__ hb) {
  __shared__ __bf16 h_sh[Bn * Hn];
  __shared__ float  c_sh[Bn * Hn];

  int dir = blockIdx.x;
  const float*  preact = dir ? preact_b : preact_f;
  const __bf16* whh    = dir ? whhb_b   : whhb_f;
  float*        hout   = dir ? hb       : hf;

  int tid  = threadIdx.x;
  int w    = tid >> 5;                  // 0..31
  int lane = tid & 31;
  int hi   = lane >> 4;

  for (int idx = tid; idx < Bn * Hn; idx += 1024) {
    h_sh[idx] = (__bf16)0.0f;
    c_sh[idx] = 0.0f;
  }
  __syncthreads();

  int n_lane = w * 16 + (lane & 15);    // 'within' column this lane owns
  const __bf16* wrow_i = whh + (0 * Hn + n_lane) * Hn;
  const __bf16* wrow_f = whh + (1 * Hn + n_lane) * Hn;
  const __bf16* wrow_g = whh + (2 * Hn + n_lane) * Hn;
  const __bf16* wrow_o = whh + (3 * Hn + n_lane) * Hn;
  const __bf16* arow   = h_sh + (lane & 15) * Hn;

  for (int t = 0; t < Ln; ++t) {
    // prefetch next step's preact block (128 KB) so it overlaps the GEMM
    if (t + 1 < Ln) {
      const float* pn = preact + (t + 1) * Bn * FH;
      __builtin_prefetch(pn + tid * 32, 0, 3);
      __builtin_prefetch(pn + tid * 32 + 16, 0, 3);
    }

    v8f ai = {0.f,0.f,0.f,0.f,0.f,0.f,0.f,0.f};
    v8f af = ai, ag = ai, ao = ai;
    if (t > 0) {                        // t==0: h==0 -> GEMM is all zeros
#pragma unroll
      for (int kk = 0; kk < Hn / 32; ++kk) {
        int k0 = kk * 32;
        v16bf a  = afrag_bf16(arow, k0, hi);
        v16bf b0 = bfrag_bf16(wrow_i, k0, hi);
        v16bf b1 = bfrag_bf16(wrow_f, k0, hi);
        v16bf b2 = bfrag_bf16(wrow_g, k0, hi);
        v16bf b3 = bfrag_bf16(wrow_o, k0, hi);
        WMMA_BF16(ai, a, b0);
        WMMA_BF16(af, a, b1);
        WMMA_BF16(ag, a, b2);
        WMMA_BF16(ao, a, b3);
      }
    }
    __syncthreads();                    // all waves done reading h_sh

    const float* pt = preact + t * Bn * FH;
#pragma unroll
    for (int g8 = 0; g8 < 8; ++g8) {
      int brow = g8 + 8 * hi;           // batch index
      int wi   = n_lane;
      float gi = ai[g8] + pt[brow * FH + 0 * Hn + wi];
      float gf = af[g8] + pt[brow * FH + 1 * Hn + wi];
      float gg = ag[g8] + pt[brow * FH + 2 * Hn + wi];
      float go = ao[g8] + pt[brow * FH + 3 * Hn + wi];
      float c  = c_sh[brow * Hn + wi];
      float si = 1.f / (1.f + __expf(-gi));
      float sf = 1.f / (1.f + __expf(-gf));
      float so = 1.f / (1.f + __expf(-go));
      float cn = sf * c + si * tanhf(gg);
      float hn = so * tanhf(cn);
      c_sh[brow * Hn + wi] = cn;
      h_sh[brow * Hn + wi] = (__bf16)hn;
      int ot = dir ? (Ln - 1 - t) : t;
      hout[(brow * Ln + ot) * Hn + wi] = hn;
    }
    __syncthreads();                    // h_sh updated before next step
  }
}

// ---------------- kernel 4: combine + oc(bf16) + emission ------------------
__global__ void combine_kernel(const float* __restrict__ hf, const float* __restrict__ hb,
                               const int* __restrict__ bio_gold,
                               const float* __restrict__ bio_emb,
                               const float* __restrict__ emission_w,
                               const float* __restrict__ emission_b,
                               __bf16* __restrict__ oc, float* __restrict__ emi) {
  int bl  = blockIdx.x;
  int tid = threadIdx.x;               // 128
  __shared__ float o_sh[Hn];
  for (int h = tid; h < Hn; h += 128) {
    float o = 0.5f * (hf[bl * Hn + h] + hb[bl * Hn + h]);
    o_sh[h] = o;
    oc[bl * OCW + h] = (__bf16)o;
  }
  int bg = bio_gold[bl];
  for (int e = tid; e < BIOE; e += 128)
    oc[bl * OCW + Hn + e] = (__bf16)bio_emb[bg * BIOE + e];
  __syncthreads();
  if (tid < 16) {
    float s = 0.f;
    if (tid < Tn) {
      s = emission_b[tid];
      for (int h = 0; h < Hn; ++h) s += o_sh[h] * emission_w[tid * Hn + h];
    }
    emi[bl * 16 + tid] = s;
  }
}

// ---------------- kernel 5: CRF NLL (1 wave, lane b handles batch b) -------
__global__ void crf_kernel(const float* __restrict__ emi, const int* __restrict__ tokens,
                           const int* __restrict__ bio_gold,
                           const float* __restrict__ trans, const float* __restrict__ start,
                           const float* __restrict__ endv, float* __restrict__ crf_out) {
  __shared__ float tr[Tn * Tn], st[Tn], en[Tn], red[Bn];
  int tid = threadIdx.x;               // 32
  for (int i = tid; i < Tn * Tn; i += 32) tr[i] = trans[i];
  if (tid < Tn) { st[tid] = start[tid]; en[tid] = endv[tid]; }
  __syncthreads();

  if (tid < Bn) {
    int b = tid;
    const float* e  = emi + b * Ln * 16;
    const int*   tg = bio_gold + b * Ln;
    const int*   tk = tokens + b * Ln;

    int prev = tg[0];
    float score = st[prev] + e[prev];
    float alpha[Tn];
    for (int j = 0; j < Tn; ++j) alpha[j] = st[j] + e[j];

    for (int t = 1; t < Ln; ++t) {
      int mt = (tk[t] != 0);
      int ct = tg[t];
      if (mt) { score += tr[prev * Tn + ct] + e[t * 16 + ct]; prev = ct; }
      float nxt[Tn];
      for (int j = 0; j < Tn; ++j) {
        float mx = -1e30f;
        for (int i = 0; i < Tn; ++i) mx = fmaxf(mx, alpha[i] + tr[i * Tn + j]);
        float s = 0.f;
        for (int i = 0; i < Tn; ++i) s += __expf(alpha[i] + tr[i * Tn + j] - mx);
        nxt[j] = mx + __logf(s) + e[t * 16 + j];
      }
      if (mt) for (int j = 0; j < Tn; ++j) alpha[j] = nxt[j];
    }
    score += en[prev];
    float mx = -1e30f;
    for (int j = 0; j < Tn; ++j) mx = fmaxf(mx, alpha[j] + en[j]);
    float s = 0.f;
    for (int j = 0; j < Tn; ++j) s += __expf(alpha[j] + en[j] - mx);
    red[b] = score - (mx + __logf(s));
  }
  __syncthreads();
  if (tid == 0) {
    float s = 0.f;
    for (int b = 0; b < Bn; ++b) s += red[b];
    crf_out[0] = -s / (float)Bn;
  }
}

// ---------------- kernel 6a: u / v projections (relu, bf16 out) ------------
__global__ void uv_proj_kernel(const __bf16* __restrict__ oc,
                               const __bf16* __restrict__ suwb, const float* __restrict__ sel_u_b,
                               const __bf16* __restrict__ svwb, const float* __restrict__ sel_v_b,
                               __bf16* __restrict__ u, __bf16* __restrict__ v) {
  int mt  = blockIdx.x;                // 0..127
  int sel = blockIdx.y;                // 0=u, 1=v
  const __bf16* W    = sel ? svwb : suwb;
  const float*  bias = sel ? sel_v_b : sel_u_b;
  __bf16*       out  = sel ? v : u;

  int w = threadIdx.x >> 5, lane = threadIdx.x & 31, hi = lane >> 4;
  const __bf16* arow = oc + (mt * 16 + (lane & 15)) * OCW;

  for (int nt2 = 0; nt2 < 2; ++nt2) {
    int nt = w * 2 + nt2;
    int n  = nt * 16 + (lane & 15);
    const __bf16* wrow = W + n * OCW;
    v8f acc = {0.f,0.f,0.f,0.f,0.f,0.f,0.f,0.f};
#pragma unroll
    for (int kk = 0; kk < OCW / 32; ++kk) {
      int k0 = kk * 32;
      v16bf a = afrag_bf16(arow, k0, hi);
      v16bf b = bfrag_bf16(wrow, k0, hi);
      WMMA_BF16(acc, a, b);
    }
    float bn = bias[n];
#pragma unroll
    for (int g = 0; g < 8; ++g) {
      int row = mt * 16 + g + 8 * hi;
      out[row * RELE + n] = (__bf16)fmaxf(acc[g] + bn, 0.f);
    }
  }
}

// ---------------- kernel 6b: P = u@w1^T + sel_uv_b ; Q = v@w2^T ------------
__global__ void pq_proj_kernel(const __bf16* __restrict__ u, const __bf16* __restrict__ v,
                               const __bf16* __restrict__ w1b, const __bf16* __restrict__ w2b,
                               const float* __restrict__ sel_uv_b,
                               float* __restrict__ P, float* __restrict__ Q) {
  int mt  = blockIdx.x;                // 0..127
  int sel = blockIdx.y;                // 0=P(u,w1,+bias), 1=Q(v,w2)
  const __bf16* A   = sel ? v : u;
  const __bf16* W   = sel ? w2b : w1b;
  float*        out = sel ? Q : P;

  int w = threadIdx.x >> 5, lane = threadIdx.x & 31, hi = lane >> 4;
  const __bf16* arow = A + (mt * 16 + (lane & 15)) * RELE;

  for (int nt2 = 0; nt2 < 2; ++nt2) {
    int nt = w * 2 + nt2;
    int n  = nt * 16 + (lane & 15);
    const __bf16* wrow = W + n * RELE;
    v8f acc = {0.f,0.f,0.f,0.f,0.f,0.f,0.f,0.f};
#pragma unroll
    for (int kk = 0; kk < RELE / 32; ++kk) {
      int k0 = kk * 32;
      v16bf a = afrag_bf16(arow, k0, hi);
      v16bf b = bfrag_bf16(wrow, k0, hi);
      WMMA_BF16(acc, a, b);
    }
    float bn = sel ? 0.f : sel_uv_b[n];
#pragma unroll
    for (int g = 0; g < 8; ++g) {
      int row = mt * 16 + g + 8 * hi;
      out[row * RELE + n] = acc[g] + bn;
    }
  }
}

// ---------------- kernel 7: fused selection einsum + BCE -------------------
// block = (b,i). logits[j,r] = sum_h relu(P[b,i,h]+Q[b,j,h]) * rel_emb[r,h]
// A-fragments built on the fly (uv never materialized); rel_emb pre-packed
// bf16 zero-padded to 64 rows so B loads are branch-free. BCE reduced here.
__global__ void selection_kernel(const float* __restrict__ P, const float* __restrict__ Q,
                                 const __bf16* __restrict__ relb,
                                 const int* __restrict__ tokens, const int* __restrict__ gold,
                                 float* __restrict__ partial) {
  int blk = blockIdx.x;                // 0..2047
  int b = blk >> 7, i = blk & 127;
  int tid = threadIdx.x;               // 256
  __shared__ float Pi[RELE];
  __shared__ int   mj[Ln];
  __shared__ float red[256];

  if (tid < RELE) {
    Pi[tid] = P[(b * Ln + i) * RELE + tid];
    mj[tid] = (tokens[b * Ln + tid] != 0) ? 1 : 0;
  }
  __syncthreads();

  float part = 0.f;
  int mask_i = (tokens[b * Ln + i] != 0);   // block-uniform
  if (mask_i) {
    int w = tid >> 5, lane = tid & 31, hi = lane >> 4;
    int j_lane = w * 16 + (lane & 15);      // A row (j) for this lane
    const float*  qrow = Q + (b * Ln + j_lane) * RELE;

    v8f accv[4];
#pragma unroll
    for (int nt = 0; nt < 4; ++nt) accv[nt] = (v8f){0.f,0.f,0.f,0.f,0.f,0.f,0.f,0.f};

#pragma unroll
    for (int kk = 0; kk < RELE / 32; ++kk) {
      int k0 = kk * 32;
      int h0 = k0 + hi * 8;
      v16bf a;
#pragma unroll
      for (int ii = 0; ii < 8; ++ii)
        a[ii] = (__bf16)fmaxf(Pi[h0 + ii] + qrow[h0 + ii], 0.f);
#pragma unroll
      for (int ii = 0; ii < 8; ++ii)
        a[8 + ii] = (__bf16)fmaxf(Pi[h0 + 16 + ii] + qrow[h0 + 16 + ii], 0.f);

#pragma unroll
      for (int nt = 0; nt < 4; ++nt) {
        int r = nt * 16 + (lane & 15);
        v16bf bb = bfrag_bf16(relb + r * RELE, k0, hi);
        WMMA_BF16(accv[nt], a, bb);
      }
    }

#pragma unroll
    for (int nt = 0; nt < 4; ++nt) {
      int r = nt * 16 + (lane & 15);
      if (r < Rn) {
#pragma unroll
        for (int g = 0; g < 8; ++g) {
          int j = w * 16 + g + 8 * hi;
          float x  = accv[nt][g];
          float gv = (float)gold[((b * Ln + i) * Rn + r) * Ln + j];
          float bce = fmaxf(x, 0.f) - x * gv + log1pf(__expf(-fabsf(x)));
          part += bce * (float)mj[j];
        }
      }
    }
  }

  red[tid] = part;
  __syncthreads();
  for (int s = 128; s > 0; s >>= 1) {
    if (tid < s) red[tid] += red[tid + s];
    __syncthreads();
  }
  if (tid == 0) partial[blk] = red[0];
}

// ---------------- kernel 8: deterministic final reduction ------------------
__global__ void final_reduce_kernel(const float* __restrict__ partial,
                                    const float* __restrict__ crf_val,
                                    const int* __restrict__ tokens,
                                    float* __restrict__ out) {
  __shared__ float red[256];
  __shared__ int   redi[256];
  int tid = threadIdx.x;
  float s = 0.f; int c = 0;
  for (int idx = tid; idx < BL; idx += 256) {
    s += partial[idx];
    c += (tokens[idx] != 0) ? 1 : 0;
  }
  red[tid] = s; redi[tid] = c;
  __syncthreads();
  for (int st = 128; st > 0; st >>= 1) {
    if (tid < st) { red[tid] += red[tid + st]; redi[tid] += redi[tid + st]; }
    __syncthreads();
  }
  if (tid == 0) out[0] = crf_val[0] + red[0] / (float)redi[0];
}

// ---------------- host launcher --------------------------------------------
extern "C" void kernel_launch(void* const* d_in, const int* in_sizes, int n_in,
                              void* d_out, int out_size, void* d_ws, size_t ws_size,
                              hipStream_t stream) {
  const int*   tokens     = (const int*)d_in[0];
  const int*   bio_gold   = (const int*)d_in[1];
  const int*   sel_gold   = (const int*)d_in[2];
  /* d_in[3] = is_train (unused) */
  const float* word_emb   = (const float*)d_in[4];
  const float* rel_emb    = (const float*)d_in[5];
  const float* bio_emb    = (const float*)d_in[6];
  const float* w_ih_f     = (const float*)d_in[7];
  const float* w_hh_f     = (const float*)d_in[8];
  const float* b_ih_f     = (const float*)d_in[9];
  const float* b_hh_f     = (const float*)d_in[10];
  const float* w_ih_b     = (const float*)d_in[11];
  const float* w_hh_b     = (const float*)d_in[12];
  const float* b_ih_b     = (const float*)d_in[13];
  const float* b_hh_b     = (const float*)d_in[14];
  const float* emission_w = (const float*)d_in[15];
  const float* emission_b = (const float*)d_in[16];
  const float* sel_u_w    = (const float*)d_in[17];
  const float* sel_u_b    = (const float*)d_in[18];
  const float* sel_v_w    = (const float*)d_in[19];
  const float* sel_v_b    = (const float*)d_in[20];
  const float* sel_uv_w   = (const float*)d_in[21];
  const float* sel_uv_b   = (const float*)d_in[22];
  const float* crf_trans  = (const float*)d_in[23];
  const float* crf_start  = (const float*)d_in[24];
  const float* crf_end    = (const float*)d_in[25];

  char* ws = (char*)d_ws;
  size_t off = 0;
  auto alloc = [&](size_t bytes) -> char* {
    char* p = ws + off;
    off = (off + bytes + 255) & ~(size_t)255;
    return p;
  };
  __bf16* emb      = (__bf16*)alloc((size_t)BL * EPn * 2);
  float*  preact_f = (float*) alloc((size_t)Ln * Bn * FH * 4);
  float*  preact_b = (float*) alloc((size_t)Ln * Bn * FH * 4);
  float*  hf       = (float*) alloc((size_t)BL * Hn * 4);
  float*  hb       = (float*) alloc((size_t)BL * Hn * 4);
  __bf16* oc       = (__bf16*)alloc((size_t)BL * OCW * 2);
  float*  emi      = (float*) alloc((size_t)BL * 16 * 4);
  __bf16* u        = (__bf16*)alloc((size_t)BL * RELE * 2);
  __bf16* v        = (__bf16*)alloc((size_t)BL * RELE * 2);
  float*  Pm       = (float*) alloc((size_t)BL * RELE * 4);
  float*  Qm       = (float*) alloc((size_t)BL * RELE * 4);
  float*  partial  = (float*) alloc((size_t)BL * 4);
  float*  crf_val  = (float*) alloc(256);
  // pre-packed bf16 weights
  __bf16* wihfb    = (__bf16*)alloc((size_t)FH * EPn * 2);
  __bf16* wihbb    = (__bf16*)alloc((size_t)FH * EPn * 2);
  __bf16* whhfb    = (__bf16*)alloc((size_t)FH * Hn * 2);
  __bf16* whhbb    = (__bf16*)alloc((size_t)FH * Hn * 2);
  __bf16* suwb     = (__bf16*)alloc((size_t)RELE * OCW * 2);
  __bf16* svwb     = (__bf16*)alloc((size_t)RELE * OCW * 2);
  __bf16* w1b      = (__bf16*)alloc((size_t)RELE * RELE * 2);
  __bf16* w2b      = (__bf16*)alloc((size_t)RELE * RELE * 2);
  __bf16* relb     = (__bf16*)alloc((size_t)RPn * RELE * 2);

  // weight packing (one-shot, independent)
  pack_bf16_kernel<<<FH,   64, 0, stream>>>(w_ih_f,   En,      0,    FH,   En,   EPn,  wihfb);
  pack_bf16_kernel<<<FH,   64, 0, stream>>>(w_ih_b,   En,      0,    FH,   En,   EPn,  wihbb);
  pack_bf16_kernel<<<FH,   64, 0, stream>>>(w_hh_f,   Hn,      0,    FH,   Hn,   Hn,   whhfb);
  pack_bf16_kernel<<<FH,   64, 0, stream>>>(w_hh_b,   Hn,      0,    FH,   Hn,   Hn,   whhbb);
  pack_bf16_kernel<<<RELE, 64, 0, stream>>>(sel_u_w,  OCW,     0,    RELE, OCW,  OCW,  suwb);
  pack_bf16_kernel<<<RELE, 64, 0, stream>>>(sel_v_w,  OCW,     0,    RELE, OCW,  OCW,  svwb);
  pack_bf16_kernel<<<RELE, 64, 0, stream>>>(sel_uv_w, 2*RELE,  0,    RELE, RELE, RELE, w1b);
  pack_bf16_kernel<<<RELE, 64, 0, stream>>>(sel_uv_w, 2*RELE,  RELE, RELE, RELE, RELE, w2b);
  pack_bf16_kernel<<<RPn,  64, 0, stream>>>(rel_emb,  RELE,    0,    Rn,   RELE, RELE, relb);

  embed_kernel<<<BL, 64, 0, stream>>>(tokens, word_emb, emb);

  input_proj_kernel<<<dim3(Ln, FH / 64), 128, 0, stream>>>(
      emb, wihfb, b_ih_f, b_hh_f, preact_f, 0);
  input_proj_kernel<<<dim3(Ln, FH / 64), 128, 0, stream>>>(
      emb, wihbb, b_ih_b, b_hh_b, preact_b, 1);

  lstm_rec_kernel<<<2, 1024, 0, stream>>>(preact_f, preact_b, whhfb, whhbb, hf, hb);

  combine_kernel<<<BL, 128, 0, stream>>>(hf, hb, bio_gold, bio_emb,
                                         emission_w, emission_b, oc, emi);

  crf_kernel<<<1, 32, 0, stream>>>(emi, tokens, bio_gold,
                                   crf_trans, crf_start, crf_end, crf_val);

  uv_proj_kernel<<<dim3(BL / 16, 2), 128, 0, stream>>>(
      oc, suwb, sel_u_b, svwb, sel_v_b, u, v);

  pq_proj_kernel<<<dim3(BL / 16, 2), 128, 0, stream>>>(
      u, v, w1b, w2b, sel_uv_b, Pm, Qm);

  selection_kernel<<<BL, 256, 0, stream>>>(Pm, Qm, relb, tokens, sel_gold, partial);

  final_reduce_kernel<<<1, 256, 0, stream>>>(partial, crf_val, tokens, (float*)d_out);

  (void)in_sizes; (void)n_in; (void)out_size; (void)ws_size;
}
